// stackCLSTM_40209483825638
// MI455X (gfx1250) — compile-verified, matrix operand order checked by reading, not run on hardware
//
#include <hip/hip_runtime.h>

typedef __bf16 bf16;
typedef __attribute__((ext_vector_type(16))) __bf16 v16bf;
typedef __attribute__((ext_vector_type(8)))  __bf16 v8bf;
typedef __attribute__((ext_vector_type(8)))  float  v8f;

#define TT 32
#define BB 128
#define II 64
#define HH 128
#define EE 4
#define IH 192        // I+H
#define CATW 768      // IH*E
#define HE 512        // H*E

#define WH_TILES_N 32 // 512/16
#define WH_TILES_K 24 // 768/32
#define G_TILES_N 8   // 128/16
#define G_TILES_K 6   // 192/32
#define WHN   (WH_TILES_N*WH_TILES_K*32*16)       // 393216 packed bf16
#define GATEN (5*EE*G_TILES_N*G_TILES_K*32*16)    // 491520 packed bf16

#define FRAG_BYTES   1024                          // 32 lanes * 32 B
#define GSTRIDE_B    ((size_t)EE*G_TILES_N*G_TILES_K*FRAG_BYTES)  // per-gate block, bytes

#define WMMA_BF16(A, Bm, C) \
  __builtin_amdgcn_wmma_f32_16x16x32_bf16(false, (A), false, (Bm), (short)0, (C), false, false)

__device__ __forceinline__ float sigmoidf_(float x) {
  return 1.0f / (1.0f + __expf(-x));
}
__device__ __forceinline__ float tanhf_(float x) {
  // tanh via single hardware exp: tanh(x) = 1 - 2/(exp(2x)+1)
  float e = __expf(2.0f * x);
  return 1.0f - 2.0f / (e + 1.0f);
}

// ---------------------------------------------------------------------------
// Prep: fp32 weights -> bf16, pre-packed into per-lane WMMA B-fragment order.
// For a K=32 x N=16 bf16 B tile, lane l supplies column n=(l&15) and K values
// kb..kb+7 (frag elems 0..7) and kb+16..kb+23 (elems 8..15), kb=(l>>4)*8
// (ISA 7.12.2 16-bit operand layout). One fragment = 32 lanes x 32 B = 1 KB.
// ---------------------------------------------------------------------------
__global__ __launch_bounds__(256) void clstm_pack(
    const float* __restrict__ Wh_w,
    const float* __restrict__ Wi_w, const float* __restrict__ Wf_w,
    const float* __restrict__ Wd_w, const float* __restrict__ Wg_w,
    const float* __restrict__ Wo_w,
    bf16* __restrict__ whP, bf16* __restrict__ gateP)
{
  int idx = blockIdx.x * blockDim.x + threadIdx.x;
  if (idx < WHN) {
    int j = idx & 15;
    int lane = (idx >> 4) & 31;
    int tt = idx >> 9;
    int kt = tt % WH_TILES_K;
    int nt = tt / WH_TILES_K;
    int n = nt * 16 + (lane & 15);
    int kb = (lane >> 4) * 8;
    int kl = (j < 8) ? (kb + j) : (kb + 16 + (j - 8));
    int k = kt * 32 + kl;
    whP[idx] = (bf16)Wh_w[k * HE + n];
  }
  if (idx < GATEN) {
    int j = idx & 15;
    int lane = (idx >> 4) & 31;
    int tt = idx >> 9;
    int kt = tt % G_TILES_K; tt /= G_TILES_K;
    int nt = tt % G_TILES_N; tt /= G_TILES_N;
    int e  = tt % EE;
    int g  = tt / EE;
    int n = nt * 16 + (lane & 15);
    int kb = (lane >> 4) * 8;
    int kl = (j < 8) ? (kb + j) : (kb + 16 + (j - 8));
    int k = kt * 32 + kl;
    const float* W = (g == 0) ? Wi_w : (g == 1) ? Wf_w : (g == 2) ? Wd_w
                   : (g == 3) ? Wg_w : Wo_w;
    gateP[idx] = (bf16)W[(e * IH + k) * HH + n];
  }
}

// ---------------------------------------------------------------------------
// Main: one block per 16 batch rows, persistent over all T steps.
// 8 waves; stage 2: each wave does 4 N-tiles of the [16,768]x[768,512] GEMM;
// stage 3: each wave owns 4 (e, n-tile) pairs and runs all 5 gate GEMMs plus
// the pointwise ODE + 101-point trapezoid quadrature for that tile.
// B-fragment loads are (per-lane base) + immediate offset; unrolling is
// bounded so only a few fragments are in flight (no VGPR spills).
// ---------------------------------------------------------------------------
__global__ __launch_bounds__(256) void clstm_main(
    const float* __restrict__ x_time, const float* __restrict__ dt_time,
    const float* __restrict__ Wh_b,
    const float* __restrict__ Wi_b, const float* __restrict__ Wf_b,
    const float* __restrict__ Wd_b, const float* __restrict__ Wg_b,
    const float* __restrict__ Wo_b,
    const float* __restrict__ alpha,
    const bf16* __restrict__ whP, const bf16* __restrict__ gateP,
    float* __restrict__ out)
{
  __shared__ __align__(32) bf16 catA[16][CATW];      // [row][ (x,h0)(x,h1).. ] bf16
  __shared__ __align__(32) bf16 ivA[EE][16][IH];     // [e][row][ x | hs ]     bf16
  __shared__ float cBuf[EE][16][HH];                 // cell state fp32
  __shared__ float lamBuf[EE][16];
  __shared__ float intBuf[EE][16];
  __shared__ float dtb[16];

  const int tid  = threadIdx.x;
  const int lane = tid & 31;
  const int wv   = tid >> 5;
  const int b0   = blockIdx.x * 16;
  const int rowA = lane & 15;
  const int kb   = (lane >> 4) * 8;

  // per-lane weight base pointers (loop-invariant; tiles become imm offsets)
  const char* whLane = (const char*)whP   + (size_t)lane * 32;
  const char* gLane  = (const char*)gateP + (size_t)lane * 32;
  // per-lane LDS A-fragment base inside a row (offsets kt*64 B are immediates)
  const bf16* aCat   = &catA[rowA][kb];

  for (int i = tid; i < 16 * CATW; i += 256) ((bf16*)catA)[i] = (bf16)0.0f;
  for (int i = tid; i < EE * 16 * HH; i += 256) ((float*)cBuf)[i] = 0.0f;
  __syncthreads();

  for (int t = 0; t < TT; ++t) {
    // ---- stage 1: stream x tile in as bf16; zero accumulators ----
    for (int i = tid; i < 16 * II; i += 256) {
      int r = i >> 6, cx = i & 63;
      bf16 v = (bf16)x_time[(size_t)(t * BB + b0 + r) * II + cx];
      #pragma unroll
      for (int m = 0; m < EE; ++m) {
        catA[r][m * IH + cx] = v;
        ivA[m][r][cx] = v;
      }
      if (t + 1 < TT)  // keep the sequential loop fed (global_prefetch_b8)
        __builtin_prefetch(&x_time[(size_t)((t + 1) * BB + b0 + r) * II + cx], 0, 1);
    }
    if (tid < 16) dtb[tid] = dt_time[t * BB + b0 + tid];
    if (tid < EE * 16) { ((float*)lamBuf)[tid] = 0.0f; ((float*)intBuf)[tid] = 0.0f; }
    __syncthreads();

    // ---- stage 2: hs = cat @ Wh_w + Wh_b  ([16,768]x[768,512]) ----
    #pragma unroll 1
    for (int nti = 0; nti < 4; ++nti) {
      const int nt = wv * 4 + nti;
      const char* pnt = whLane + (size_t)nt * (WH_TILES_K * FRAG_BYTES);
      v8f acc = {0.f,0.f,0.f,0.f,0.f,0.f,0.f,0.f};
      #pragma unroll 4
      for (int kt = 0; kt < WH_TILES_K; ++kt) {
        union { v16bf v; v8bf h[2]; } a;
        a.h[0] = *(const v8bf*)(aCat + kt * 32);
        a.h[1] = *(const v8bf*)(aCat + kt * 32 + 16);
        v16bf bm = *(const v16bf*)(pnt + (size_t)kt * FRAG_BYTES);
        acc = WMMA_BF16(a.v, bm, acc);
      }
      const int n = nt * 16 + (lane & 15);
      const float bias = Wh_b[n];
      const int e = n >> 7, hc = n & 127;   // column -> (e, h)
      #pragma unroll
      for (int r = 0; r < 8; ++r) {
        int rr = r + 8 * (lane >> 4);
        ivA[e][rr][II + hc] = (bf16)(acc[r] + bias);
      }
    }
    __syncthreads();

    // ---- stage 3: 5 gate GEMMs per (e, n-tile) pair + pointwise ODE ----
    #pragma unroll 1
    for (int pi = 0; pi < 4; ++pi) {
      const int p = wv * 4 + pi;
      const int e = p >> 3, nt = p & 7;
      const char* pg   = gLane + (size_t)((e * G_TILES_N + nt) * G_TILES_K) * FRAG_BYTES;
      const bf16* aIv  = &ivA[e][rowA][kb];
      v8f ai = {0.f,0.f,0.f,0.f,0.f,0.f,0.f,0.f};
      v8f af = ai, ad = ai, ag = ai, ao = ai;
      #pragma unroll 2
      for (int kt = 0; kt < G_TILES_K; ++kt) {
        union { v16bf v; v8bf h[2]; } a;
        a.h[0] = *(const v8bf*)(aIv + kt * 32);
        a.h[1] = *(const v8bf*)(aIv + kt * 32 + 16);
        // all five gate fragments: same per-lane base, immediate offsets
        v16bf bi = *(const v16bf*)(pg + (size_t)kt * FRAG_BYTES + 0 * GSTRIDE_B);
        v16bf bf = *(const v16bf*)(pg + (size_t)kt * FRAG_BYTES + 1 * GSTRIDE_B);
        v16bf bd = *(const v16bf*)(pg + (size_t)kt * FRAG_BYTES + 2 * GSTRIDE_B);
        v16bf bg = *(const v16bf*)(pg + (size_t)kt * FRAG_BYTES + 3 * GSTRIDE_B);
        v16bf bo = *(const v16bf*)(pg + (size_t)kt * FRAG_BYTES + 4 * GSTRIDE_B);
        ai = WMMA_BF16(a.v, bi, ai);
        af = WMMA_BF16(a.v, bf, af);
        ad = WMMA_BF16(a.v, bd, ad);
        ag = WMMA_BF16(a.v, bg, ag);
        ao = WMMA_BF16(a.v, bo, ao);
      }
      const int col = nt * 16 + (lane & 15);
      const float al  = alpha[e * HH + col];
      const float bi_ = Wi_b[e * HH + col], bf_ = Wf_b[e * HH + col];
      const float bd_ = Wd_b[e * HH + col], bg_ = Wg_b[e * HH + col];
      const float bo_ = Wo_b[e * HH + col];
      #pragma unroll
      for (int r = 0; r < 8; ++r) {
        const int rr = r + 8 * (lane >> 4);
        const float dtv   = dtb[rr];
        const float cprev = cBuf[e][rr][col];
        const float ig = sigmoidf_(ai[r] + bi_);
        const float fg = sigmoidf_(af[r] + bf_);
        const float gg = fg * cprev + ig * tanhf_(ag[r] + bg_);
        const float dl = __expf(ad[r] + bd_);
        const float og = sigmoidf_(ao[r] + bo_);
        const float diff = cprev - gg;
        const float cn = gg + diff * __expf(-dl * dtv);
        const float hn = og * tanhf_(cn);
        cBuf[e][rr][col] = cn;
        catA[rr][e * IH + II + col] = (bf16)hn;      // h feeds next step's cat
        atomicAdd(&lamBuf[e][rr], al * hn);
        // trapezoid over 101 points; exp(-dl*ts_p) via geometric recurrence
        const float ratio = __expf(-dl * dtv * 0.01f);
        float ef = 1.0f, s = 0.0f, f0 = 0.0f, fl = 0.0f;
        #pragma unroll 4
        for (int pp = 0; pp <= 100; ++pp) {
          float tv = tanhf_(gg + diff * ef);
          s += tv;
          if (pp == 0) f0 = tv;
          fl = tv;
          ef *= ratio;
        }
        s -= 0.5f * (f0 + fl);
        atomicAdd(&intBuf[e][rr], al * og * s * dtv * 0.01f);
      }
    }
    __syncthreads();

    // ---- stage 4: emit lambda / integral for this step ----
    if (tid < EE * 16) {
      int e = tid & 3, rr = tid >> 2;
      int b = b0 + rr;
      out[((size_t)t * BB + b) * EE + e] = lamBuf[e][rr];
      out[(size_t)TT * BB * EE + ((size_t)t * BB + b) * EE + e] = intBuf[e][rr];
    }
    __syncthreads();
  }
}

extern "C" void kernel_launch(void* const* d_in, const int* in_sizes, int n_in,
                              void* d_out, int out_size, void* d_ws, size_t ws_size,
                              hipStream_t stream) {
  const float* x_time  = (const float*)d_in[0];
  const float* dt_time = (const float*)d_in[1];
  const float* Wh_w = (const float*)d_in[2];
  const float* Wh_b = (const float*)d_in[3];
  const float* Wi_w = (const float*)d_in[4];
  const float* Wi_b = (const float*)d_in[5];
  const float* Wf_w = (const float*)d_in[6];
  const float* Wf_b = (const float*)d_in[7];
  const float* Wd_w = (const float*)d_in[8];
  const float* Wd_b = (const float*)d_in[9];
  const float* Wg_w = (const float*)d_in[10];
  const float* Wg_b = (const float*)d_in[11];
  const float* Wo_w = (const float*)d_in[12];
  const float* Wo_b = (const float*)d_in[13];
  const float* alpha = (const float*)d_in[14];

  // workspace: bf16 weights in pre-packed WMMA fragment order (~1.69 MB)
  bf16* whP   = reinterpret_cast<bf16*>(d_ws);
  bf16* gateP = reinterpret_cast<bf16*>((char*)d_ws + (size_t)WHN * sizeof(bf16));

  clstm_pack<<<(GATEN + 255) / 256, 256, 0, stream>>>(
      Wh_w, Wi_w, Wf_w, Wd_w, Wg_w, Wo_w, whP, gateP);

  clstm_main<<<BB / 16, 256, 0, stream>>>(
      x_time, dt_time, Wh_b, Wi_b, Wf_b, Wd_b, Wg_b, Wo_b,
      alpha, whP, gateP, (float*)d_out);
}
